// GREcA_89120571392346
// MI455X (gfx1250) — matrix operand order repaired and checked by reading, not compiled, not run on hardware
//
#include <hip/hip_runtime.h>
#include <math.h>

// Problem constants (match reference)
#define NUM_ITEMS  10000
#define NUM_USERS  10000
#define NUM_GROUPS 5000
#define EMB_DIM    64
#define BATCH      16384

#define NCOLS 80            // 64 emb cols + 1 ones col (row count) + 15 zero pad
#define NT    5             // NCOLS / 16 WMMA N-tiles
#define KT    313           // ceil(10000 / 32) K-tiles (both GEMMs have K = 10000)
#define KSPLIT 4            // deterministic split-K chunks

typedef __bf16 v16bf __attribute__((ext_vector_type(16)));
typedef float  v8f   __attribute__((ext_vector_type(8)));
typedef float  v4f   __attribute__((ext_vector_type(4)));   // clang vector: OK for nontemporal builtins

// CDNA5 v_wmma_f32_16x16x32_bf16 per-lane K mapping (ISA 7.12.2):
// A (16x32): lanes 0-15 hold row M=lane, K in {0..7, 16..23}; lanes 16-31: K in {8..15, 24..31}
__device__ __forceinline__ int kmapA(int lane, int i) {
    return ((i >> 3) << 4) + ((lane >> 4) << 3) + (i & 7);
}
// B (32x16): lanes 0-15 hold col N=lane, K in {0..7, 16..23}; lanes 16-31: K in {8..15, 24..31}
__device__ __forceinline__ int kmapB(int lane, int i) {
    return ((i >> 3) << 4) + ((lane >> 4) << 3) + (i & 7);
}

// ---------------------------------------------------------------------------
// Pack B1 = [item_emb | ones | 0-pad] (10000x80 fp32 -> bf16) into the exact
// per-lane WMMA operand layout: Bp[(kt*NT + nt)*32 + lane][0..15].
// ---------------------------------------------------------------------------
__global__ void __launch_bounds__(128) pack_item_B(const float* __restrict__ item_emb,
                                                   __bf16* __restrict__ Bp) {
    int w    = (blockIdx.x * blockDim.x + threadIdx.x) >> 5;
    int lane = threadIdx.x & 31;
    if (w >= KT * NT) return;
    int kt = w / NT, nt = w % NT;
    int n  = nt * 16 + (lane & 15);
    __bf16* out = Bp + ((size_t)w * 32 + lane) * 16;
#pragma unroll
    for (int i = 0; i < 16; ++i) {
        int k = kt * 32 + kmapB(lane, i);
        float v = 0.0f;
        if (k < NUM_ITEMS) {
            if (n < EMB_DIM)        v = item_emb[(size_t)k * EMB_DIM + n];
            else if (n == EMB_DIM)  v = 1.0f;   // ones column -> row counts for free
        }
        out[i] = (__bf16)v;
    }
}

// ---------------------------------------------------------------------------
// Main GEMM: C_part[chunk] = A_chunk @ Bpacked.  A is fp32 {0,1} (lossless
// bf16 convert in-flight, non-temporal: streamed once).  Each wave: 16 rows x
// 80 cols, K-steps of 32 via 5x v_wmma_f32_16x16x32_bf16.  Inner loop is a
// 2-deep register-double-buffered software pipeline so the next K-step's
// A (HBM) and B (L2) loads fly under the current step's WMMAs.
// Deterministic split-K: plain stores to per-chunk partials, no atomics.
// ---------------------------------------------------------------------------
__global__ void __launch_bounds__(128) gemm_bin_bf16(
        const float*  __restrict__ A,      // [M][K] fp32
        const __bf16* __restrict__ Bp,     // packed [(KT*NT)*32][16]
        float*        __restrict__ Cpart,  // [KSPLIT][M][NCOLS] f32 partials
        int M, int K, int kt_per_chunk) {
    int wave = threadIdx.x >> 5;
    int lane = threadIdx.x & 31;
    int tile = blockIdx.x * 4 + wave;      // 16-row tile index
    int row0 = tile * 16;
    if (row0 >= M) return;                 // wave-uniform: EXEC stays all-1s for WMMA

    int half = lane >> 4;                  // which K half this lane holds for A
    int row  = row0 + (lane & 15);
    bool rvalid = row < M;
    float sel = rvalid ? 1.0f : 0.0f;      // zero out-of-range rows after clamped load
    const float* arow  = A + (size_t)(rvalid ? row : (M - 1)) * K;
    const float* abase = arow + half * 8;  // lane's A stream: K = kt*32 + half*8 + {0..7,16..23}

    int ktn = (K + 31) / 32;
    int kt0 = blockIdx.y * kt_per_chunk;
    int kt1 = kt0 + kt_per_chunk; if (kt1 > ktn) kt1 = ktn;
    bool tailTile = (kt1 * 32 > K);        // this chunk owns the ragged last K-tile
    int  ktFull   = tailTile ? kt1 - 1 : kt1;

    v8f c0 = {}, c1 = {}, c2 = {}, c3 = {}, c4 = {};

    auto loadAraw = [&](int kt, v4f& f0, v4f& f1, v4f& f2, v4f& f3) {
        const v4f* p = (const v4f*)(abase + kt * 32);
        f0 = __builtin_nontemporal_load(p);       // NT: 600 MB stream, don't pollute L2
        f1 = __builtin_nontemporal_load(p + 1);
        f2 = __builtin_nontemporal_load(p + 4);
        f3 = __builtin_nontemporal_load(p + 5);
    };
    auto cvtA = [&](v4f f0, v4f f1, v4f f2, v4f f3) -> v16bf {
        v16bf a;
        a[0]=(__bf16)(f0.x*sel);  a[1]=(__bf16)(f0.y*sel);  a[2]=(__bf16)(f0.z*sel);  a[3]=(__bf16)(f0.w*sel);
        a[4]=(__bf16)(f1.x*sel);  a[5]=(__bf16)(f1.y*sel);  a[6]=(__bf16)(f1.z*sel);  a[7]=(__bf16)(f1.w*sel);
        a[8]=(__bf16)(f2.x*sel);  a[9]=(__bf16)(f2.y*sel);  a[10]=(__bf16)(f2.z*sel); a[11]=(__bf16)(f2.w*sel);
        a[12]=(__bf16)(f3.x*sel); a[13]=(__bf16)(f3.y*sel); a[14]=(__bf16)(f3.z*sel); a[15]=(__bf16)(f3.w*sel);
        return a;
    };
    auto loadB = [&](int kt, v16bf* b) {
        const v16bf* bp = (const v16bf*)(Bp + ((size_t)kt * NT * 32 + lane) * 16);
#pragma unroll
        for (int nt = 0; nt < NT; ++nt) b[nt] = bp[nt * 32];   // 32 v16bf = 512 bf16 per N-tile
    };
    auto mma = [&](v16bf a, const v16bf* b) {
        c0 = __builtin_amdgcn_wmma_f32_16x16x32_bf16(false, a, false, b[0], (short)0, c0, false, false);
        c1 = __builtin_amdgcn_wmma_f32_16x16x32_bf16(false, a, false, b[1], (short)0, c1, false, false);
        c2 = __builtin_amdgcn_wmma_f32_16x16x32_bf16(false, a, false, b[2], (short)0, c2, false, false);
        c3 = __builtin_amdgcn_wmma_f32_16x16x32_bf16(false, a, false, b[3], (short)0, c3, false, false);
        c4 = __builtin_amdgcn_wmma_f32_16x16x32_bf16(false, a, false, b[4], (short)0, c4, false, false);
    };

    // ---- software-pipelined main loop: even/odd register double buffers ----
    if (kt0 < ktFull) {
        v4f e0, e1, e2, e3;  v16bf eb[NT];
        v4f o0, o1, o2, o3;  v16bf ob[NT];
        loadAraw(kt0, e0, e1, e2, e3);
        loadB(kt0, eb);
        for (int kt = kt0; kt < ktFull; kt += 2) {
            bool hasOdd = (kt + 1 < ktFull);             // uniform
            if (hasOdd) { loadAraw(kt + 1, o0, o1, o2, o3); loadB(kt + 1, ob); }
            mma(cvtA(e0, e1, e2, e3), eb);
            if (hasOdd) {
                if (kt + 2 < ktFull) { loadAraw(kt + 2, e0, e1, e2, e3); loadB(kt + 2, eb); }
                mma(cvtA(o0, o1, o2, o3), ob);
            }
        }
    }
    // ---- ragged last K-tile (10000 % 32 != 0): guarded scalar loads ----
    if (tailTile) {
        int kb = (kt1 - 1) * 32;
        v16bf a;
#pragma unroll
        for (int i = 0; i < 16; ++i) {
            int k = kb + kmapA(lane, i);
            float v = (k < K) ? arow[k] * sel : 0.0f;
            a[i] = (__bf16)v;
        }
        v16bf tb[NT];
        loadB(kt1 - 1, tb);
        mma(a, tb);
    }

    // C layout: VGPR v -> row row0 + v + (lane<16 ? 0 : 8), col = nt*16 + (lane&15)
    int crow0 = row0 + half * 8;
    int ccol  = lane & 15;
    float* cbase = Cpart + (size_t)blockIdx.y * M * NCOLS;
#pragma unroll
    for (int v = 0; v < 8; ++v) {
        int rr = crow0 + v;
        if (rr < M) {
            float* dst = cbase + (size_t)rr * NCOLS + ccol;
            dst[0]  = c0[v];
            dst[16] = c1[v];
            dst[32] = c2[v];
            dst[48] = c3[v];
            dst[64] = c4[v];
        }
    }
}

// ---------------------------------------------------------------------------
// Finalize GEMM1: sum split-K partials, divide by max(count,1), and write the
// result DIRECTLY as the packed bf16 B operand for GEMM2 (ones col appended).
// ---------------------------------------------------------------------------
__global__ void __launch_bounds__(128) pack_user_B(const float* __restrict__ C1p,
                                                   __bf16* __restrict__ Bp) {
    int w    = (blockIdx.x * blockDim.x + threadIdx.x) >> 5;
    int lane = threadIdx.x & 31;
    if (w >= KT * NT) return;
    int kt = w / NT, nt = w % NT;
    int n  = nt * 16 + (lane & 15);
    __bf16* out = Bp + ((size_t)w * 32 + lane) * 16;
#pragma unroll
    for (int i = 0; i < 16; ++i) {
        int u = kt * 32 + kmapB(lane, i);
        float v = 0.0f;
        if (u < NUM_USERS) {
            if (n < EMB_DIM) {
                float cnt = 0.0f, val = 0.0f;
#pragma unroll
                for (int ch = 0; ch < KSPLIT; ++ch) {
                    const float* row = C1p + ((size_t)ch * NUM_USERS + u) * NCOLS;
                    cnt += row[EMB_DIM];
                    val += row[n];
                }
                v = val / fmaxf(cnt, 1.0f);     // clamped divisor (reference semantics)
            } else if (n == EMB_DIM) v = 1.0f;
        }
        out[i] = (__bf16)v;
    }
}

// Finalize GEMM2: group_embeds = sum(partials)/sum(count)  (raw divisor, no clamp)
__global__ void finalize_groups(const float* __restrict__ C2p, float* __restrict__ Gemb) {
    int idx = blockIdx.x * blockDim.x + threadIdx.x;
    if (idx >= NUM_GROUPS * EMB_DIM) return;
    int g = idx >> 6, d = idx & 63;
    float cnt = 0.0f, val = 0.0f;
#pragma unroll
    for (int ch = 0; ch < KSPLIT; ++ch) {
        const float* row = C2p + ((size_t)ch * NUM_GROUPS + g) * NCOLS;
        cnt += row[EMB_DIM];
        val += row[d];
    }
    Gemb[idx] = val / cnt;
}

// ---------------------------------------------------------------------------
// Gather + tiny MLP: sigmoid(relu(concat(g,e) @ W1 + b1) @ W2 + b2).  W1 is
// 4 KB (L0/L2 hot); one thread per sample is plenty for 34 MFLOP.
// ---------------------------------------------------------------------------
__global__ void mlp_predict(const float* __restrict__ Gemb, const float* __restrict__ item_emb,
                            const float* __restrict__ W1, const float* __restrict__ b1,
                            const float* __restrict__ W2, const float* __restrict__ b2,
                            const int* __restrict__ gidx, const int* __restrict__ iidx,
                            float* __restrict__ out) {
    int i = blockIdx.x * blockDim.x + threadIdx.x;
    if (i >= BATCH) return;
    const float* ge = Gemb     + (size_t)gidx[i] * EMB_DIM;
    const float* ie = item_emb + (size_t)iidx[i] * EMB_DIM;
    float h[8];
#pragma unroll
    for (int j = 0; j < 8; ++j) h[j] = b1[j];
    for (int d = 0; d < EMB_DIM; ++d) {
        float x = ge[d];
#pragma unroll
        for (int j = 0; j < 8; ++j) h[j] = fmaf(x, W1[d * 8 + j], h[j]);
    }
    for (int d = 0; d < EMB_DIM; ++d) {
        float x = ie[d];
#pragma unroll
        for (int j = 0; j < 8; ++j) h[j] = fmaf(x, W1[(EMB_DIM + d) * 8 + j], h[j]);
    }
    float z = b2[0];
#pragma unroll
    for (int j = 0; j < 8; ++j) z = fmaf(fmaxf(h[j], 0.0f), W2[j], z);
    out[i] = 1.0f / (1.0f + __expf(-z));
}

// ---------------------------------------------------------------------------
extern "C" void kernel_launch(void* const* d_in, const int* in_sizes, int n_in,
                              void* d_out, int out_size, void* d_ws, size_t ws_size,
                              hipStream_t stream) {
    const float* item_emb = (const float*)d_in[0];
    const float* ui       = (const float*)d_in[1];
    const float* gu       = (const float*)d_in[2];
    const float* W1       = (const float*)d_in[3];
    const float* b1       = (const float*)d_in[4];
    const float* W2       = (const float*)d_in[5];
    const float* b2       = (const float*)d_in[6];
    const int*   gidx     = (const int*)d_in[7];
    const int*   iidx     = (const int*)d_in[8];
    float* out = (float*)d_out;

    // Workspace carve-up (~23.7 MB), all regions fully overwritten every call.
    char*  ws  = (char*)d_ws;
    size_t oC1 = 0;
    size_t oC2 = oC1 + (size_t)KSPLIT * NUM_USERS  * NCOLS * sizeof(float);
    size_t oB1 = oC2 + (size_t)KSPLIT * NUM_GROUPS * NCOLS * sizeof(float);
    size_t oB2 = oB1 + (size_t)KT * NT * 32 * 16 * sizeof(__bf16);
    size_t oG  = oB2 + (size_t)KT * NT * 32 * 16 * sizeof(__bf16);
    size_t need = oG + (size_t)NUM_GROUPS * EMB_DIM * sizeof(float);
    if (ws_size < need) return;
    float*  C1p  = (float*)(ws + oC1);
    float*  C2p  = (float*)(ws + oC2);
    __bf16* B1   = (__bf16*)(ws + oB1);
    __bf16* B2   = (__bf16*)(ws + oB2);
    float*  Gemb = (float*)(ws + oG);

    int packBlocks = (KT * NT * 32 + 127) / 128;
    int ktpc = (KT + KSPLIT - 1) / KSPLIT;   // K-tiles per split-K chunk

    // 1) pack item_emb -> WMMA-layout bf16 B1
    pack_item_B<<<packBlocks, 128, 0, stream>>>(item_emb, B1);
    // 2) user_embeds partials = ui @ B1  (streams 400 MB of ui once)
    dim3 g1((NUM_USERS + 63) / 64, KSPLIT);
    gemm_bin_bf16<<<g1, 128, 0, stream>>>(ui, B1, C1p, NUM_USERS, NUM_ITEMS, ktpc);
    // 3) reduce partials, divide by clamped count, re-pack as bf16 B2
    pack_user_B<<<packBlocks, 128, 0, stream>>>(C1p, B2);
    // 4) group_embeds partials = gu @ B2  (streams 200 MB of gu once)
    dim3 g2((NUM_GROUPS + 63) / 64, KSPLIT);
    gemm_bin_bf16<<<g2, 128, 0, stream>>>(gu, B2, C2p, NUM_GROUPS, NUM_USERS, ktpc);
    // 5) reduce + divide -> group_embeds fp32
    finalize_groups<<<(NUM_GROUPS * EMB_DIM + 255) / 256, 256, 0, stream>>>(C2p, Gemb);
    // 6) gather + MLP + sigmoid -> d_out [16384]
    mlp_predict<<<(BATCH + 255) / 256, 256, 0, stream>>>(Gemb, item_emb, W1, b1, W2, b2,
                                                         gidx, iidx, out);
}
